// StateSpaceModel_87308095193484
// MI455X (gfx1250) — matrix operand order repaired and checked by reading, not compiled, not run on hardware
//
#include <hip/hip_runtime.h>

#define B_  4096
#define L_  512
#define A_  64
#define HD_ 1024
#define H_  32

typedef __attribute__((ext_vector_type(16))) __bf16 v16bf;
typedef __attribute__((ext_vector_type(8)))  __bf16 v8bf;
typedef __attribute__((ext_vector_type(8)))  float  v8f;

union V16U { v16bf v; v8bf h[2]; };

// A-fragment (16x32 bf16, row-major activations). Contiguous k order used here (not the ISA table order):
// lanes 0-15 : M=lane,    VGPR0-7 -> K = 0..15 (contiguous, matches B)
// lanes 16-31: M=lane-16, VGPR0-7 -> K = 16..31
__device__ __forceinline__ v16bf load_a32(const __bf16* rowp, int k0, int grp) {
    V16U u;
    u.h[0] = *(const v8bf*)(rowp + k0 + grp * 16);
    u.h[1] = *(const v8bf*)(rowp + k0 + grp * 16 + 8);
    return u.v;
}

__device__ __forceinline__ float gelu_exact(float x) {
    return 0.5f * x * (1.0f + erff(x * 0.70710678118654752440f));
}

#define WMMA_BF16(a, b, c) \
    __builtin_amdgcn_wmma_f32_16x16x32_bf16(false, (a), false, (b), (short)0, (c), false, false)

// Double-buffered 2M x 4N GEMM core: acc[2][4] += A[2 tiles] x B[4 tiles].
// Peeled final iteration -> steady-state loop has constant-stride addressing.
template <int K, int LDA>
__device__ __forceinline__ void gemm_2x4(
    const __bf16* __restrict__ arow0, const __bf16* __restrict__ arow1,
    const __bf16* const brow[4], int grp, v8f acc[2][4])
{
    v16bf aC0 = load_a32(arow0, 0, grp);
    v16bf aC1 = load_a32(arow1, 0, grp);
    v16bf bC[4];
    #pragma unroll
    for (int t = 0; t < 4; ++t) bC[t] = *(const v16bf*)(brow[t]);

    #pragma unroll 2
    for (int k0 = 32; k0 < K; k0 += 32) {
        __builtin_prefetch(arow0 + k0 + 224, 0, 3);
        v16bf aN0 = load_a32(arow0, k0, grp);
        v16bf aN1 = load_a32(arow1, k0, grp);
        v16bf bN[4];
        #pragma unroll
        for (int t = 0; t < 4; ++t) bN[t] = *(const v16bf*)(brow[t] + k0);

        #pragma unroll
        for (int t = 0; t < 4; ++t) {
            acc[0][t] = WMMA_BF16(aC0, bC[t], acc[0][t]);
            acc[1][t] = WMMA_BF16(aC1, bC[t], acc[1][t]);
        }
        aC0 = aN0; aC1 = aN1;
        #pragma unroll
        for (int t = 0; t < 4; ++t) bC[t] = bN[t];
    }
    // peeled tail: consume last preloaded fragments
    #pragma unroll
    for (int t = 0; t < 4; ++t) {
        acc[0][t] = WMMA_BF16(aC0, bC[t], acc[0][t]);
        acc[1][t] = WMMA_BF16(aC1, bC[t], acc[1][t]);
    }
}

// ---------------------------------------------------------------------------
// K1/K2: out = gelu(LN(A @ W + bias) * gamma + beta), N = 1024 fixed.
// A: [4096 x K] bf16 row-major (stride == K). WT: [1024 x K] bf16 (transposed).
// Block = 32 rows x 1024 cols; 512 thr = 16 waves, each 2 M-tiles x 4 N-tiles.
// ---------------------------------------------------------------------------
template <int K>
__global__ __launch_bounds__(512) void mlp_ln_gelu(
    const __bf16* __restrict__ Aact, const __bf16* __restrict__ BT,
    const float* __restrict__ bias, const float* __restrict__ gamma,
    const float* __restrict__ beta, __bf16* __restrict__ out)
{
    const int tid  = threadIdx.x;
    const int wave = tid >> 5;          // 0..15
    const int lane = tid & 31;
    const int grp  = lane >> 4;
    const int l15  = lane & 15;
    const int m0   = blockIdx.x << 5;   // 32 rows per block
    const int nb   = wave * 64;         // 4 N-tiles of 16

    const __bf16* arow0 = Aact + (size_t)(m0 + l15) * K;
    const __bf16* arow1 = Aact + (size_t)(m0 + 16 + l15) * K;
    const __bf16* brow[4];
    #pragma unroll
    for (int t = 0; t < 4; ++t)
        brow[t] = BT + (size_t)(nb + t * 16 + l15) * K + grp * 16;

    v8f acc[2][4] = {};
    gemm_2x4<K, K>(arow0, arow1, brow, grp, acc);

    __shared__ float sSum[32];
    __shared__ float sSq[32];
    if (tid < 32) { sSum[tid] = 0.0f; sSq[tid] = 0.0f; }
    __syncthreads();

    // bake bias into accumulators (pre-activation values)
    #pragma unroll
    for (int t = 0; t < 4; ++t) {
        float bv = bias[nb + t * 16 + l15];
        #pragma unroll
        for (int mi = 0; mi < 2; ++mi)
            #pragma unroll
            for (int r = 0; r < 8; ++r) acc[mi][t][r] += bv;
    }

    // per-row partials: lane holds rows m = mi*16 + grp*8 + r, 4 cols each
    #pragma unroll
    for (int mi = 0; mi < 2; ++mi) {
        float s[8], q[8];
        #pragma unroll
        for (int r = 0; r < 8; ++r) {
            s[r] = 0.0f; q[r] = 0.0f;
            #pragma unroll
            for (int t = 0; t < 4; ++t) { float x = acc[mi][t][r]; s[r] += x; q[r] += x * x; }
            #pragma unroll
            for (int mk = 8; mk >= 1; mk >>= 1) {
                s[r] += __shfl_xor(s[r], mk, 16);
                q[r] += __shfl_xor(q[r], mk, 16);
            }
        }
        if (l15 == 0) {
            #pragma unroll
            for (int r = 0; r < 8; ++r) {
                atomicAdd(&sSum[mi * 16 + grp * 8 + r], s[r]);
                atomicAdd(&sSq [mi * 16 + grp * 8 + r], q[r]);
            }
        }
    }
    __syncthreads();

    #pragma unroll
    for (int mi = 0; mi < 2; ++mi) {
        #pragma unroll
        for (int r = 0; r < 8; ++r) {
            const int m = mi * 16 + grp * 8 + r;
            float mean = sSum[m] * (1.0f / 1024.0f);
            float var  = sSq [m] * (1.0f / 1024.0f) - mean * mean;
            float inv  = rsqrtf(var + 1e-5f);
            const size_t rowg = (size_t)(m0 + m) * 1024;
            #pragma unroll
            for (int t = 0; t < 4; ++t) {
                int n = nb + t * 16 + l15;
                float x = (acc[mi][t][r] - mean) * inv * gamma[n] + beta[n];
                out[rowg + n] = (__bf16)gelu_exact(x);
            }
        }
    }
}

// ---------------------------------------------------------------------------
// K3: dz = h2 @ W3 + b3; z_next = z + sigmoid(gate)*dz
//     -> z (f32, in place), traj output, za[:, :512] (bf16 for next step / K4)
// Block = 32 rows x 512 cols; 256 thr = 8 waves, each 2 M-tiles x 4 N-tiles.
// ---------------------------------------------------------------------------
__global__ __launch_bounds__(256) void dz_update(
    const __bf16* __restrict__ h2, const __bf16* __restrict__ W3T,
    const float* __restrict__ b3, float* __restrict__ z,
    __bf16* __restrict__ za, float* __restrict__ traj,
    const float* __restrict__ gate, int step)
{
    const int tid  = threadIdx.x;
    const int wave = tid >> 5;          // 0..7
    const int lane = tid & 31;
    const int grp  = lane >> 4;
    const int l15  = lane & 15;
    const int m0   = blockIdx.x << 5;
    const int nb   = wave * 64;

    const __bf16* arow0 = h2 + (size_t)(m0 + l15) * 1024;
    const __bf16* arow1 = h2 + (size_t)(m0 + 16 + l15) * 1024;
    const __bf16* brow[4];
    #pragma unroll
    for (int t = 0; t < 4; ++t)
        brow[t] = W3T + (size_t)(nb + t * 16 + l15) * 1024 + grp * 16;

    v8f acc[2][4] = {};
    gemm_2x4<1024, 1024>(arow0, arow1, brow, grp, acc);

    const float gs = 1.0f / (1.0f + expf(-gate[0]));
    #pragma unroll
    for (int t = 0; t < 4; ++t) {
        const int n  = nb + t * 16 + l15;
        const float bv = b3[n];
        #pragma unroll
        for (int mi = 0; mi < 2; ++mi) {
            #pragma unroll
            for (int r = 0; r < 8; ++r) {
                const int row = m0 + mi * 16 + grp * 8 + r;
                const size_t zi = (size_t)row * 512 + n;
                float zn = z[zi] + gs * (acc[mi][t][r] + bv);
                z[zi] = zn;
                traj[((size_t)row * H_ + step) * L_ + n] = zn;
                za[(size_t)row * 576 + n] = (__bf16)zn;
            }
        }
    }
}

// ---------------------------------------------------------------------------
// K4: u = gelu(z_next @ uW1 + ub1); sigma = softplus(u @ uW2 + ub2)
// A = za[:, :512] (stride 576). Block = 32 rows; 8 waves, 2 M x 4 N tiles.
// ---------------------------------------------------------------------------
__global__ __launch_bounds__(256) void sigma_kernel(
    const __bf16* __restrict__ za, const __bf16* __restrict__ uW1T,
    const float* __restrict__ ub1, const float* __restrict__ uW2,
    const float* __restrict__ ub2, float* __restrict__ sigs, int step)
{
    const int tid  = threadIdx.x;
    const int wave = tid >> 5;          // 0..7
    const int lane = tid & 31;
    const int grp  = lane >> 4;
    const int l15  = lane & 15;
    const int m0   = blockIdx.x << 5;
    const int nb   = wave * 64;

    const __bf16* arow0 = za + (size_t)(m0 + l15) * 576;
    const __bf16* arow1 = za + (size_t)(m0 + 16 + l15) * 576;
    const __bf16* brow[4];
    #pragma unroll
    for (int t = 0; t < 4; ++t)
        brow[t] = uW1T + (size_t)(nb + t * 16 + l15) * 512 + grp * 16;

    v8f acc[2][4] = {};
    gemm_2x4<512, 576>(arow0, arow1, brow, grp, acc);

    float p[2][8] = {};
    #pragma unroll
    for (int t = 0; t < 4; ++t) {
        const int n = nb + t * 16 + l15;
        const float w2 = uW2[n];
        const float bv = ub1[n];
        #pragma unroll
        for (int mi = 0; mi < 2; ++mi)
            #pragma unroll
            for (int r = 0; r < 8; ++r)
                p[mi][r] += gelu_exact(acc[mi][t][r] + bv) * w2;
    }
    #pragma unroll
    for (int mi = 0; mi < 2; ++mi)
        #pragma unroll
        for (int r = 0; r < 8; ++r)
            #pragma unroll
            for (int mk = 8; mk >= 1; mk >>= 1)
                p[mi][r] += __shfl_xor(p[mi][r], mk, 16);

    __shared__ float sRed[8][32];
    if (l15 == 0) {
        #pragma unroll
        for (int mi = 0; mi < 2; ++mi)
            #pragma unroll
            for (int r = 0; r < 8; ++r)
                sRed[wave][mi * 16 + grp * 8 + r] = p[mi][r];
    }
    __syncthreads();

    if (tid < 32) {
        float sum = 0.0f;
        #pragma unroll
        for (int w = 0; w < 8; ++w) sum += sRed[w][tid];
        float x  = sum + ub2[0];
        float sp = (x > 20.0f) ? x : log1pf(expf(x));
        sigs[(size_t)(m0 + tid) * H_ + step] = sp;
    }
}

// ------------------------- prep / glue kernels -----------------------------
__global__ void wconv_t(const float* __restrict__ W, __bf16* __restrict__ WT,
                        int K, int N)
{
    int idx = blockIdx.x * 256 + threadIdx.x;   // grid sized exactly K*N/256
    int k = idx / N, n = idx % N;
    WT[(size_t)n * K + k] = (__bf16)W[idx];
}

__global__ void init_z(const float* __restrict__ z0, float* __restrict__ z,
                       __bf16* __restrict__ za)
{
    int idx = blockIdx.x * 256 + threadIdx.x;   // B*512 threads
    int row = idx >> 9, c = idx & 511;
    float v = z0[idx];
    z[idx] = v;
    za[(size_t)row * 576 + c] = (__bf16)v;
}

__global__ void act_conv(const float* __restrict__ actions,
                         __bf16* __restrict__ za, int step)
{
    int idx = blockIdx.x * 256 + threadIdx.x;   // B*64 threads
    int row = idx >> 6, j = idx & 63;
    za[(size_t)row * 576 + 512 + j] =
        (__bf16)actions[((size_t)row * H_ + step) * A_ + j];
}

// ---------------------------------------------------------------------------
extern "C" void kernel_launch(void* const* d_in, const int* in_sizes, int n_in,
                              void* d_out, int out_size, void* d_ws, size_t ws_size,
                              hipStream_t stream)
{
    (void)in_sizes; (void)n_in; (void)out_size; (void)ws_size;

    const float* z0      = (const float*)d_in[0];
    const float* actions = (const float*)d_in[1];
    const float* W1  = (const float*)d_in[2];
    const float* b1  = (const float*)d_in[3];
    const float* g1  = (const float*)d_in[4];
    const float* be1 = (const float*)d_in[5];
    const float* W2  = (const float*)d_in[6];
    const float* b2  = (const float*)d_in[7];
    const float* g2  = (const float*)d_in[8];
    const float* be2 = (const float*)d_in[9];
    const float* W3  = (const float*)d_in[10];
    const float* b3  = (const float*)d_in[11];
    const float* uW1 = (const float*)d_in[12];
    const float* ub1 = (const float*)d_in[13];
    const float* uW2 = (const float*)d_in[14];
    const float* ub2 = (const float*)d_in[15];
    const float* gate = (const float*)d_in[16];

    char* ws = (char*)d_ws;
    size_t off = 0;
    auto alloc = [&](size_t bytes) -> char* {
        char* p = ws + off;
        off += (bytes + 255) & ~(size_t)255;
        return p;
    };
    __bf16* W1T  = (__bf16*)alloc((size_t)576 * 1024 * 2);
    __bf16* W2T  = (__bf16*)alloc((size_t)1024 * 1024 * 2);
    __bf16* W3T  = (__bf16*)alloc((size_t)1024 * 512 * 2);
    __bf16* uW1T = (__bf16*)alloc((size_t)512 * 512 * 2);
    float*  z    = (float*) alloc((size_t)B_ * 512 * 4);
    __bf16* za   = (__bf16*)alloc((size_t)B_ * 576 * 2);
    __bf16* h1   = (__bf16*)alloc((size_t)B_ * 1024 * 2);
    __bf16* h2   = (__bf16*)alloc((size_t)B_ * 1024 * 2);

    float* traj = (float*)d_out;                      // [B, H, L]
    float* sigs = traj + (size_t)B_ * H_ * L_;        // [B, H, 1]

    // one-time weight transpose + bf16 conversion
    wconv_t<<<(576 * 1024) / 256, 256, 0, stream>>>(W1, W1T, 576, 1024);
    wconv_t<<<(1024 * 1024) / 256, 256, 0, stream>>>(W2, W2T, 1024, 1024);
    wconv_t<<<(1024 * 512) / 256, 256, 0, stream>>>(W3, W3T, 1024, 512);
    wconv_t<<<(512 * 512) / 256, 256, 0, stream>>>(uW1, uW1T, 512, 512);
    init_z<<<(B_ * 512) / 256, 256, 0, stream>>>(z0, z, za);

    for (int t = 0; t < H_; ++t) {
        act_conv<<<(B_ * A_) / 256, 256, 0, stream>>>(actions, za, t);
        mlp_ln_gelu<576><<<B_ / 32, 512, 0, stream>>>(za, W1T, b1, g1, be1, h1);
        mlp_ln_gelu<1024><<<B_ / 32, 512, 0, stream>>>(h1, W2T, b2, g2, be2, h2);
        dz_update<<<B_ / 32, 256, 0, stream>>>(h2, W3T, b3, z, za, traj, gate, t);
        sigma_kernel<<<B_ / 32, 256, 0, stream>>>(za, uW1T, ub1, uW2, ub2, sigs, t);
    }
}
